// MultiHeadAttention_12300786336369
// MI455X (gfx1250) — compile-verified, matrix operand order checked by reading, not run on hardware
//
#include <hip/hip_runtime.h>
#include <hip/hip_bf16.h>

typedef __attribute__((ext_vector_type(16))) _Float16 v16h;
typedef __attribute__((ext_vector_type(8)))  _Float16 v8h;
typedef __attribute__((ext_vector_type(8)))  float    v8f;

#define S_LEN  4096
#define DMODEL 512
#define HEADS  8
#define HD     64
#define MROWS  8192   // N * S_LEN

// ---------------------------------------------------------------------------
// 16x32 f16 operand tile in CDNA5 WMMA A/B register layout.
// Lane L (L<16): row = base+L, halves = {k0+0..7, k0+16..23}
// Lane L+16    : row = base+L, halves = {k0+8..15, k0+24..31}
// ---------------------------------------------------------------------------
__device__ __forceinline__ v16h load_tile_k(const _Float16* src, int row_base,
                                            int ld, int k0) {
  const int lane = threadIdx.x & 31;
  const int hs   = lane >> 4;
  const _Float16* p = src + (size_t)(row_base + (lane & 15)) * ld + k0 + hs * 8;
  v8h lo = *(const v8h*)(p);
  v8h hi = *(const v8h*)(p + 16);
  v16h r;
#pragma unroll
  for (int i = 0; i < 8; ++i) { r[i] = lo[i]; r[i + 8] = hi[i]; }
  return r;
}

__device__ __forceinline__ v8f wmma_f16(v16h a, v16h b, v8f c) {
  return __builtin_amdgcn_wmma_f32_16x16x32_f16(false, a, false, b,
                                                (short)0, c, false, false);
}

// ---------------------------------------------------------------------------
// DPP16 cross-lane butterfly within each 16-lane half (pure VALU, no DS).
// ---------------------------------------------------------------------------
template <int CTRL>
__device__ __forceinline__ float dpp_movf(float x) {
  return __int_as_float(__builtin_amdgcn_update_dpp(
      0, __float_as_int(x), CTRL, 0xF, 0xF, true));
}

__device__ __forceinline__ float red16_max(float x) {
  x = fmaxf(x, dpp_movf<0x0B1>(x));  // quad_perm xor1
  x = fmaxf(x, dpp_movf<0x04E>(x));  // quad_perm xor2
  x = fmaxf(x, dpp_movf<0x141>(x));  // row_half_mirror
  x = fmaxf(x, dpp_movf<0x140>(x));  // row_mirror
  return x;
}

__device__ __forceinline__ float red16_sum(float x) {
  x += dpp_movf<0x0B1>(x);
  x += dpp_movf<0x04E>(x);
  x += dpp_movf<0x141>(x);
  x += dpp_movf<0x140>(x);
  return x;
}

// ---------------------------------------------------------------------------
// Elementwise f32 -> f16 convert
// ---------------------------------------------------------------------------
__global__ __launch_bounds__(256) void cvt_f32_f16(const float* __restrict__ in,
                                                   _Float16* __restrict__ out,
                                                   int n) {
  int i = blockIdx.x * blockDim.x + threadIdx.x;
  if (i < n) out[i] = (_Float16)in[i];
}

// ---------------------------------------------------------------------------
// Projection GEMM: C[m,n] = (sum_k X[m,k]*W[n,k] + bias[n]) * oscale
// M=8192, N=512, K=512. Block 256 (8 waves), wave tile 32x64, block 128x128.
// MODE 0: f16 per-head row-major  [((b*8+h)*4096+s)*64 + d]
// MODE 1: f16 per-head transposed [((b*8+h)*64+d)*4096 + s]
// MODE 2: f32 row-major out[m*512 + n]
// ---------------------------------------------------------------------------
template <int MODE>
__global__ __launch_bounds__(256) void proj_gemm(
    const _Float16* __restrict__ X, const _Float16* __restrict__ W,
    const float* __restrict__ bias, void* __restrict__ out, float oscale) {
  const int w    = threadIdx.x >> 5;
  const int lane = threadIdx.x & 31;
  const int hs   = lane >> 4;
  const int ln   = lane & 15;
  const int wm   = w >> 1;
  const int wn   = w & 1;
  const int m0   = blockIdx.x * 128 + wm * 32;
  const int n0   = blockIdx.y * 128 + wn * 64;

  v8f c[2][4];
#pragma unroll
  for (int i = 0; i < 2; ++i)
#pragma unroll
    for (int j = 0; j < 4; ++j) c[i][j] = {};

  for (int k0 = 0; k0 < DMODEL; k0 += 32) {
    v16h a0 = load_tile_k(X, m0,      DMODEL, k0);
    v16h a1 = load_tile_k(X, m0 + 16, DMODEL, k0);
#pragma unroll
    for (int j = 0; j < 4; ++j) {
      v16h b = load_tile_k(W, n0 + 16 * j, DMODEL, k0);
      c[0][j] = wmma_f16(a0, b, c[0][j]);
      c[1][j] = wmma_f16(a1, b, c[1][j]);
    }
  }

#pragma unroll
  for (int i = 0; i < 2; ++i) {
#pragma unroll
    for (int j = 0; j < 4; ++j) {
      const int n = n0 + 16 * j + ln;
      const float bn = bias[n];
#pragma unroll
      for (int v = 0; v < 8; ++v) {
        const int m = m0 + 16 * i + hs * 8 + v;
        const float val = (c[i][j][v] + bn) * oscale;
        if (MODE == 2) {
          ((float*)out)[(size_t)m * DMODEL + n] = val;
        } else {
          const int b = m >> 12;
          const int s = m & (S_LEN - 1);
          const int h = n >> 6;
          const int d = n & (HD - 1);
          _Float16* o = (_Float16*)out;
          if (MODE == 0)
            o[(((size_t)b * HEADS + h) * S_LEN + s) * HD + d] = (_Float16)val;
          else
            o[(((size_t)b * HEADS + h) * HD + d) * S_LEN + s] = (_Float16)val;
        }
      }
    }
  }
}

// ---------------------------------------------------------------------------
// Flash attention, base-2 online softmax (scale*log2e folded into Q).
// Grid (32,16), block 256 (8 waves). Wave: 16 q-rows x 64 d. 64-key steps:
//   8 WMMAs S=QK^T, DPP16 row reductions, exp2, P->LDS->A-layout,
//   8 WMMAs O+=PV with rescaled f32 accumulators.
// ---------------------------------------------------------------------------
__global__ __launch_bounds__(256) void flash_attn(
    const _Float16* __restrict__ Qh, const _Float16* __restrict__ Kh,
    const _Float16* __restrict__ Vt, _Float16* __restrict__ Oc) {
  const int w    = threadIdx.x >> 5;
  const int lane = threadIdx.x & 31;
  const int hs   = lane >> 4;
  const int ln   = lane & 15;
  const int bh   = blockIdx.y;
  const int b    = bh >> 3;
  const int h    = bh & 7;
  const int q0   = blockIdx.x * 128 + w * 16;

  const _Float16* Q = Qh + (size_t)bh * S_LEN * HD;
  const _Float16* K = Kh + (size_t)bh * S_LEN * HD;
  const _Float16* V = Vt + (size_t)bh * HD * S_LEN;  // [d][s]

  __shared__ _Float16 Pl[8][16][64];   // per-wave P tile, 16 KB

  const v16h aq0 = load_tile_k(Q, q0, HD, 0);
  const v16h aq1 = load_tile_k(Q, q0, HD, 32);

  v8f o[4];
#pragma unroll
  for (int j = 0; j < 4; ++j) o[j] = {};
  float m_r[8], l_r[8];
#pragma unroll
  for (int v = 0; v < 8; ++v) { m_r[v] = -__builtin_inff(); l_r[v] = 0.f; }

  for (int kt = 0; kt < S_LEN; kt += 64) {
    // S = Q K^T for 64 keys (four 16-key subtiles), d-accumulated
    v8f s[4];
#pragma unroll
    for (int t = 0; t < 4; ++t) {
      s[t] = {};
      v16h bk = load_tile_k(K, kt + 16 * t, HD, 0);
      s[t] = wmma_f16(aq0, bk, s[t]);
      bk = load_tile_k(K, kt + 16 * t, HD, 32);
      s[t] = wmma_f16(aq1, bk, s[t]);
    }

    float al[8];
#pragma unroll
    for (int v = 0; v < 8; ++v) {
      const float x0 = s[0][v], x1 = s[1][v], x2 = s[2][v], x3 = s[3][v];
      const float t = red16_max(fmaxf(fmaxf(x0, x1), fmaxf(x2, x3)));
      const float nm = fmaxf(m_r[v], t);
      al[v] = exp2f(m_r[v] - nm);
      m_r[v] = nm;

      const float p0 = exp2f(x0 - nm);
      const float p1 = exp2f(x1 - nm);
      const float p2 = exp2f(x2 - nm);
      const float p3 = exp2f(x3 - nm);
      const float rs = red16_sum((p0 + p1) + (p2 + p3));
      l_r[v] = l_r[v] * al[v] + rs;

      const int r = hs * 8 + v;
      Pl[w][r][ln]      = (_Float16)p0;
      Pl[w][r][16 + ln] = (_Float16)p1;
      Pl[w][r][32 + ln] = (_Float16)p2;
      Pl[w][r][48 + ln] = (_Float16)p3;
    }

#pragma unroll
    for (int j = 0; j < 4; ++j)
#pragma unroll
      for (int v = 0; v < 8; ++v) o[j][v] *= al[v];

    const v16h ap0 = load_tile_k(&Pl[w][0][0], 0, 64, 0);
    const v16h ap1 = load_tile_k(&Pl[w][0][0], 0, 64, 32);

#pragma unroll
    for (int j = 0; j < 4; ++j) {
      v16h bv = load_tile_k(V, 16 * j, S_LEN, kt);
      o[j] = wmma_f16(ap0, bv, o[j]);
      bv = load_tile_k(V, 16 * j, S_LEN, kt + 32);
      o[j] = wmma_f16(ap1, bv, o[j]);
    }
  }

  float inv[8];
#pragma unroll
  for (int v = 0; v < 8; ++v) inv[v] = 1.0f / l_r[v];

#pragma unroll
  for (int j = 0; j < 4; ++j) {
#pragma unroll
    for (int v = 0; v < 8; ++v) {
      const int q   = q0 + hs * 8 + v;
      const int col = h * HD + 16 * j + ln;
      Oc[((size_t)b * S_LEN + q) * DMODEL + col] = (_Float16)(o[j][v] * inv[v]);
    }
  }
}

// ---------------------------------------------------------------------------
// Host side
// ---------------------------------------------------------------------------
extern "C" void kernel_launch(void* const* d_in, const int* in_sizes, int n_in,
                              void* d_out, int out_size, void* d_ws,
                              size_t ws_size, hipStream_t stream) {
  const float* q  = (const float*)d_in[0];
  const float* k  = (const float*)d_in[1];
  const float* v  = (const float*)d_in[2];
  const float* Wq = (const float*)d_in[3];
  const float* bq = (const float*)d_in[4];
  const float* Wk = (const float*)d_in[5];
  const float* bk = (const float*)d_in[6];
  const float* Wv = (const float*)d_in[7];
  const float* bv = (const float*)d_in[8];
  const float* Wo = (const float*)d_in[9];
  const float* bo = (const float*)d_in[10];

  const size_t ACT = (size_t)MROWS * DMODEL;
  const size_t WSZ = (size_t)DMODEL * DMODEL;
  char* ws = (char*)d_ws;
  _Float16* Xq  = (_Float16*)(ws + 0 * ACT * 2);
  _Float16* Xk  = (_Float16*)(ws + 1 * ACT * 2);
  _Float16* Xv  = (_Float16*)(ws + 2 * ACT * 2);
  _Float16* Qh  = (_Float16*)(ws + 3 * ACT * 2);
  _Float16* Kh  = (_Float16*)(ws + 4 * ACT * 2);
  _Float16* Vt  = (_Float16*)(ws + 5 * ACT * 2);
  _Float16* Oc  = (_Float16*)(ws + 6 * ACT * 2);
  _Float16* Wqh = (_Float16*)(ws + 7 * ACT * 2 + 0 * WSZ * 2);
  _Float16* Wkh = (_Float16*)(ws + 7 * ACT * 2 + 1 * WSZ * 2);
  _Float16* Wvh = (_Float16*)(ws + 7 * ACT * 2 + 2 * WSZ * 2);
  _Float16* Woh = (_Float16*)(ws + 7 * ACT * 2 + 3 * WSZ * 2);

  const int cb = 256;
  cvt_f32_f16<<<(int)((ACT + cb - 1) / cb), cb, 0, stream>>>(q, Xq, (int)ACT);
  cvt_f32_f16<<<(int)((ACT + cb - 1) / cb), cb, 0, stream>>>(k, Xk, (int)ACT);
  cvt_f32_f16<<<(int)((ACT + cb - 1) / cb), cb, 0, stream>>>(v, Xv, (int)ACT);
  cvt_f32_f16<<<(int)((WSZ + cb - 1) / cb), cb, 0, stream>>>(Wq, Wqh, (int)WSZ);
  cvt_f32_f16<<<(int)((WSZ + cb - 1) / cb), cb, 0, stream>>>(Wk, Wkh, (int)WSZ);
  cvt_f32_f16<<<(int)((WSZ + cb - 1) / cb), cb, 0, stream>>>(Wv, Wvh, (int)WSZ);
  cvt_f32_f16<<<(int)((WSZ + cb - 1) / cb), cb, 0, stream>>>(Wo, Woh, (int)WSZ);

  // fold softmax scale (1/sqrt(64)) and log2(e) into Q so the attention
  // kernel runs a pure base-2 online softmax with no per-element scaling.
  const float SCL_Q = 0.125f * 1.44269504088896340736f;

  dim3 pg(MROWS / 128, DMODEL / 128);
  proj_gemm<0><<<pg, 256, 0, stream>>>(Xq, Wqh, bq, (void*)Qh, SCL_Q);
  proj_gemm<0><<<pg, 256, 0, stream>>>(Xk, Wkh, bk, (void*)Kh, 1.0f);
  proj_gemm<1><<<pg, 256, 0, stream>>>(Xv, Wvh, bv, (void*)Vt, 1.0f);

  dim3 ag(S_LEN / 128, 2 * HEADS);
  flash_attn<<<ag, 256, 0, stream>>>(Qh, Kh, Vt, Oc);

  proj_gemm<2><<<pg, 256, 0, stream>>>(Oc, Woh, bo, d_out, 1.0f);
}
